// MultiProtoAsWCos_30872224924139
// MI455X (gfx1250) — compile-verified, matrix operand order checked by reading, not run on hardware
//
#include <hip/hip_runtime.h>
#include <hip/hip_bf16.h>
#include <math.h>

typedef __attribute__((ext_vector_type(16))) _Float16 v16h;
typedef __attribute__((ext_vector_type(8)))  _Float16 v8h;
typedef __attribute__((ext_vector_type(8)))  float    v8f;
typedef __attribute__((ext_vector_type(4)))  int      v4i;
typedef __attribute__((address_space(3)))    v4i      lds_v4i;

#define C_CH   256
#define H_DIM  128
#define W_DIM  128
#define M_TOT  1280      // 5 * 16 * 16 prototypes
#define NPIX   16384     // 128*128
#define EPSN   1e-4f
#define CHUNKS (M_TOT / 16)
#define BSTRIDE 264      // LDS row stride in halfs (256 + 8 pad -> 4-bank rotate/row)
#define WAVES  8         // waves per block in the GEMM kernel

// -------------------------------------------------------------------------
// Kernel 1: 8x8 avg-pool sup_y -> 1280 scores; mask with any() fallbacks.
// -------------------------------------------------------------------------
__global__ void mask_kernel(const float* __restrict__ sup_y,
                            float* __restrict__ mask) {
    __shared__ float yv[M_TOT];
    __shared__ int any1, any2;
    int t = threadIdx.x;
    if (t == 0) { any1 = 0; any2 = 0; }
    __syncthreads();
    for (int idx = t; idx < M_TOT; idx += blockDim.x) {
        int s = idx >> 8;
        int p = idx & 255;
        int gy = p >> 4, gx = p & 15;
        float acc = 0.f;
        for (int dy = 0; dy < 8; ++dy)
            for (int dx = 0; dx < 8; ++dx)
                acc += sup_y[((size_t)s * H_DIM + gy * 8 + dy) * W_DIM + gx * 8 + dx];
        float v = acc * (1.f / 64.f);
        yv[idx] = v;
        if (v > 0.5f) atomicOr(&any1, 1);
        if (v > 0.1f) atomicOr(&any2, 1);
    }
    __syncthreads();
    for (int idx = t; idx < M_TOT; idx += blockDim.x) {
        float v = yv[idx];
        float mk;
        if (any1)      mk = (v > 0.5f) ? 1.f : 0.f;
        else if (any2) mk = (v > 0.1f) ? 1.f : 0.f;
        else           mk = 1.f;
        mask[idx] = mk;
    }
}

// -------------------------------------------------------------------------
// Kernel 2: pool sup_x, center over channels, L2-normalize, f16 proN[m][c].
// -------------------------------------------------------------------------
__global__ void proto_kernel(const float* __restrict__ sup_x,
                             _Float16* __restrict__ proN) {
    __shared__ float red[C_CH];
    int m = blockIdx.x;
    int c = threadIdx.x;
    int s = m >> 8;
    int p = m & 255;
    int gy = p >> 4, gx = p & 15;
    const float* base = sup_x + (((size_t)(s * C_CH + c)) * H_DIM + gy * 8) * W_DIM + gx * 8;
    float acc = 0.f;
    for (int dy = 0; dy < 8; ++dy)
        for (int dx = 0; dx < 8; ++dx)
            acc += base[dy * W_DIM + dx];
    float v = acc * (1.f / 64.f);

    red[c] = v; __syncthreads();
    for (int off = 128; off > 0; off >>= 1) {
        if (c < off) red[c] += red[c + off];
        __syncthreads();
    }
    float mean = red[0] * (1.f / 256.f);
    __syncthreads();
    float cv = v - mean;
    red[c] = cv * cv; __syncthreads();
    for (int off = 128; off > 0; off >>= 1) {
        if (c < off) red[c] += red[c + off];
        __syncthreads();
    }
    float nrm = fmaxf(sqrtf(red[0]), EPSN);
    proN[(size_t)m * C_CH + c] = (_Float16)(cv / nrm);
}

// -------------------------------------------------------------------------
// Kernel 3: per-pixel center+normalize query, f16 aQ[pixel][c].
// -------------------------------------------------------------------------
__global__ void qnorm_kernel(const float* __restrict__ qry,
                             _Float16* __restrict__ aQ) {
    int pix = blockIdx.x * blockDim.x + threadIdx.x;
    float sum = 0.f;
    for (int c = 0; c < C_CH; ++c) sum += qry[(size_t)c * NPIX + pix];
    float mean = sum * (1.f / 256.f);
    float ss = 0.f;
    for (int c = 0; c < C_CH; ++c) {
        float d = qry[(size_t)c * NPIX + pix] - mean;
        ss += d * d;
    }
    float inv = 1.f / fmaxf(sqrtf(ss), EPSN);
    for (int c = 0; c < C_CH; ++c)
        aQ[(size_t)pix * C_CH + c] = (_Float16)((qry[(size_t)c * NPIX + pix] - mean) * inv);
}

// -------------------------------------------------------------------------
// CDNA5 async global->LDS copy (ASYNCcnt); signature probe-confirmed:
//   (v4i* src, __attribute__((address_space(3))) v4i* dst, imm off, imm cpol)
// -------------------------------------------------------------------------
__device__ __forceinline__ void async_copy_b128(const _Float16* g, _Float16* l) {
#if __has_builtin(__builtin_amdgcn_global_load_async_to_lds_b128)
    __builtin_amdgcn_global_load_async_to_lds_b128(
        (v4i*)g, (lds_v4i*)l, 0, 0);
#else
    asm volatile("global_load_async_to_lds_b128 %0, %1, off"
                 :: "v"((unsigned)(size_t)l), "v"(g) : "memory");
#endif
}

__device__ __forceinline__ void wait_asynccnt0() {
#if __has_builtin(__builtin_amdgcn_s_wait_asynccnt)
    __builtin_amdgcn_s_wait_asynccnt(0);
#else
    asm volatile("s_wait_asynccnt 0" ::: "memory");
#endif
}

// Stage one 16x256 f16 B-chunk (8KB) into padded LDS tile: 512 16B units,
// 2 per thread across the 256-thread block.
__device__ __forceinline__ void stage_chunk(const _Float16* __restrict__ proN,
                                            int jb, _Float16* buf, int t) {
#pragma unroll
    for (int r = 0; r < 2; ++r) {
        int u   = t + r * 256;         // 0..511
        int row = u >> 5;              // 0..15
        int col = (u & 31) * 8;        // halfs, 16B units
        async_copy_b128(proN + (size_t)(jb + row) * C_CH + col,
                        buf + row * BSTRIDE + col);
    }
}

// -------------------------------------------------------------------------
// Kernel 4: WMMA f16 GEMM + masked softmax-weighted reduction.
// dists bounded in [-20,20] (unit vectors) -> exp() cannot overflow, so
// masked softmax == w = mask*exp(d); accumulate Sum(w), Sum(w*d) per lane,
// single 16-lane merge at the end. B chunks double-buffered in LDS via
// async loads; A resident in registers. All 8 B fragments are preloaded
// per chunk so ds_load latency overlaps the WMMA pipeline instead of
// serializing on s_wait_dscnt 0.
// Fragment K-map per ISA: elems 0..7 -> K=kb*32+hs*8..+7, elems 8..15 -> +16.
// -------------------------------------------------------------------------
__device__ __forceinline__ v16h load_frag(const _Float16* rowbase, int kbase) {
    v8h lo = *(const v8h*)(rowbase + kbase);
    v8h hi = *(const v8h*)(rowbase + kbase + 16);
    v16h f;
#pragma unroll
    for (int e = 0; e < 8; ++e) { f[e] = lo[e]; f[e + 8] = hi[e]; }
    return f;
}

__global__ void __launch_bounds__(32 * WAVES)
gemm_softmax_kernel(const _Float16* __restrict__ aQ,
                    const _Float16* __restrict__ proN,
                    const float* __restrict__ mask,
                    float* __restrict__ out) {
    __shared__ _Float16 sB[2][16 * BSTRIDE];   // double-buffered B chunk

    int t    = threadIdx.x;
    int wave = t >> 5;
    int lane = t & 31;
    int hs   = lane >> 4;
    int ln   = lane & 15;
    int tile = blockIdx.x * WAVES + wave;      // 16-pixel tile per wave

    // A fragments: 16 pixels x 256 channels resident in registers.
    const _Float16* arow = aQ + (size_t)(tile * 16 + ln) * C_CH;
    v16h a[8];
#pragma unroll
    for (int kb = 0; kb < 8; ++kb)
        a[kb] = load_frag(arow, kb * 32 + hs * 8);

    float se[8], sw[8];
#pragma unroll
    for (int i = 0; i < 8; ++i) { se[i] = 0.f; sw[i] = 0.f; }

    stage_chunk(proN, 0, sB[0], t);

    for (int j = 0; j < CHUNKS; ++j) {
        wait_asynccnt0();          // my staging loads done
        __syncthreads();           // everyone's data visible; prev buf free
        if (j + 1 < CHUNKS)
            stage_chunk(proN, (j + 1) * 16, sB[(j + 1) & 1], t);

        float mk = mask[j * 16 + ln];   // hoisted: latency hidden by WMMAs

        // Preload all 8 B fragments (16 ds_load_b128 in one clause), then
        // run the 8-deep WMMA chain.
        const _Float16* brow = &sB[j & 1][ln * BSTRIDE];
        v16h b[8];
#pragma unroll
        for (int kb = 0; kb < 8; ++kb)
            b[kb] = load_frag(brow, kb * 32 + hs * 8);

        v8f cacc = {};
#pragma unroll
        for (int kb = 0; kb < 8; ++kb)
            cacc = __builtin_amdgcn_wmma_f32_16x16x32_f16(
                false, a[kb], false, b[kb], (short)0, cacc, false, false);

#pragma unroll
        for (int i = 0; i < 8; ++i) {
            float d = 20.f * cacc[i];
            float w = mk * __expf(d);   // exact masked-softmax weight
            se[i] += w;
            sw[i] += w * d;
        }
    }

    // One-time 16-lane merge (xor<16 stays inside each wave half).
#pragma unroll
    for (int i = 0; i < 8; ++i) {
        float s = se[i], w = sw[i];
        s += __shfl_xor(s, 1, 32);  w += __shfl_xor(w, 1, 32);
        s += __shfl_xor(s, 2, 32);  w += __shfl_xor(w, 2, 32);
        s += __shfl_xor(s, 4, 32);  w += __shfl_xor(w, 4, 32);
        s += __shfl_xor(s, 8, 32);  w += __shfl_xor(w, 8, 32);
        if (ln == 0)
            out[tile * 16 + hs * 8 + i] = w / s;
    }
}

// -------------------------------------------------------------------------
extern "C" void kernel_launch(void* const* d_in, const int* in_sizes, int n_in,
                              void* d_out, int out_size, void* d_ws, size_t ws_size,
                              hipStream_t stream) {
    const float* qry   = (const float*)d_in[0];   // (256,128,128)
    const float* sup_x = (const float*)d_in[1];   // (5,256,128,128)
    const float* sup_y = (const float*)d_in[2];   // (5,128,128)
    float* out = (float*)d_out;                   // (128,128)

    char* ws = (char*)d_ws;
    float*    mask = (float*)ws;                                          // 5 KB
    _Float16* proN = (_Float16*)(ws + 8192);                              // 640 KB
    _Float16* aQ   = (_Float16*)(ws + 8192 + (size_t)M_TOT * C_CH * 2);   // 8 MB

    mask_kernel <<<1,          256, 0, stream>>>(sup_y, mask);
    proto_kernel<<<M_TOT,      C_CH, 0, stream>>>(sup_x, proN);
    qnorm_kernel<<<NPIX / 256, 256, 0, stream>>>(qry, aQ);
    gemm_softmax_kernel<<<NPIX / (16 * WAVES), 32 * WAVES, 0, stream>>>(aQ, proN, mask, out);
}
